// MDN_81836306858381
// MI455X (gfx1250) — matrix operand order repaired and checked by reading, not compiled, not run on hardware
//
#include <hip/hip_runtime.h>
#include <hip/hip_bf16.h>

// ---------------------------------------------------------------------------
// MDN forward, fused: h = tanh(x@W1+b1); out320 = h @ [Wpi|Wmu|Wsig] + bias;
// softmax(pi), mu passthrough, sigma exp/tanh packing.
//
// MI455X plan:
//  - bf16 WMMA (v_wmma_f32_16x16x32_bf16), f32 accumulate
//  - fully fused: [B,1024] h intermediate never touches HBM
//  - prep kernel pre-converts weights to bf16, k-major, into d_ws
//  - BM=128 rows/block, 512 threads (16 waves)
//  - weights for chunk ch+1 prefetched through VGPRs while ch computes
//  - paired WMMAs per ds-wait (shared B in stage 1, B-pairs in stage 2)
// ---------------------------------------------------------------------------

typedef __attribute__((ext_vector_type(16))) __bf16 v16bf;
typedef __attribute__((ext_vector_type(8)))  float  v8f;

union Frag16 { uint4 q[2]; v16bf v; };

__device__ __forceinline__ unsigned short f2bf(float f) {
  unsigned u = __float_as_uint(f);
  unsigned r = u + 0x7FFFu + ((u >> 16) & 1u);   // round-to-nearest-even
  return (unsigned short)(r >> 16);
}

// A-matrix fragment (16x32 bf16), row-major LDS tile [row][k], stride in ushorts.
// Lane l<16: M=l, K = {0..7, 16..23}; lane>=16: M=l-16, K = {8..15, 24..31}.
__device__ __forceinline__ v16bf load_a(const unsigned short* base, int stride,
                                        int r0, int k0, int lane) {
  int half = lane >> 4, m = lane & 15;
  const unsigned short* p = base + (r0 + m) * stride + k0 + half * 8;
  Frag16 f;
  f.q[0] = *(const uint4*)(p);
  f.q[1] = *(const uint4*)(p + 16);
  return f.v;
}

// B-matrix fragment (32x16 bf16), k-major LDS tile [col][k], stride in ushorts.
// Lane l<16: N=l, K = {0..7, 16..23}; lane>=16: N=l-16, K = {8..15, 24..31}.
__device__ __forceinline__ v16bf load_b(const unsigned short* base, int stride,
                                        int c0, int k0, int lane) {
  int half = lane >> 4, n = lane & 15;
  const unsigned short* p = base + (c0 + n) * stride + k0 + half * 8;
  Frag16 f;
  f.q[0] = *(const uint4*)(p);
  f.q[1] = *(const uint4*)(p + 16);
  return f.v;
}

__device__ __forceinline__ v8f wmma_bf16(v16bf a, v16bf b, v8f c) {
  return __builtin_amdgcn_wmma_f32_16x16x32_bf16(
      /*neg_a=*/false, a, /*neg_b=*/false, b,
      /*c_mod=*/(short)0, c, /*reuse_a=*/false, /*reuse_b=*/false);
}

// Problem constants
#define BIN   256      // input features
#define BH    1024     // hidden
#define NK    32       // mixture components
#define NCOL  320      // 32 + 96 + 192
#define BM    128      // batch rows per block
#define NT    512      // threads per block (16 waves)
#define HC    64       // hidden chunk
#define NCH   (BH / HC)

// d_ws layout (ushorts): W1t [1024 cols][256 k] then Wht [320 cols][1024 k]
#define WS_W1T_ELEMS (BH * BIN)    // 262144
#define WS_WHT_ELEMS (NCOL * BH)   // 327680
#define WS_TOTAL     (WS_W1T_ELEMS + WS_WHT_ELEMS)

// LDS geometry (ushort strides padded for bank-conflict-free b128 reads)
#define XS_STRIDE 264      // 256 + 8
#define W1_STRIDE 264
#define WH_STRIDE 72       // 64 + 8
#define HB_STRIDE 72
// pool: w1c(33792) + whc(46080) + hbuf(18432) = 98304 >= stage 64x324 f32 (82944)
#define POOL_BYTES 98304

// ---------------------------------------------------------------------------
// prep: weights f32 -> bf16, transposed to k-major-per-column, into d_ws
// ---------------------------------------------------------------------------
__global__ __launch_bounds__(256)
void mdn_prep_kernel(const float* __restrict__ W1,
                     const float* __restrict__ Wpi,
                     const float* __restrict__ Wmu,
                     const float* __restrict__ Wsig,
                     unsigned short* __restrict__ ws) {
  unsigned short* W1t = ws;                  // [1024][256]
  unsigned short* Wht = ws + WS_W1T_ELEMS;   // [320][1024]
  int i = blockIdx.x * 256 + threadIdx.x;
  if (i < WS_W1T_ELEMS) {
    int c = i >> 8, k = i & 255;             // W1 is [256 k][1024 c]
    W1t[i] = f2bf(W1[(size_t)k * BH + c]);
  } else if (i < WS_TOTAL) {
    int j = i - WS_W1T_ELEMS;
    int c = j >> 10, k = j & 1023;           // heads are [1024 k][cols]
    float v;
    if (c < NK)            v = Wpi[(size_t)k * NK + c];
    else if (c < NK + 96)  v = Wmu[(size_t)k * 96 + (c - NK)];
    else                   v = Wsig[(size_t)k * 192 + (c - NK - 96)];
    Wht[j] = f2bf(v);
  }
}

// ---------------------------------------------------------------------------
// main fused kernel
// ---------------------------------------------------------------------------
__global__ __launch_bounds__(NT)
void mdn_fused_kernel(const float* __restrict__ x,
                      const unsigned short* __restrict__ W1t,
                      const unsigned short* __restrict__ Wht,
                      const float* __restrict__ b1,
                      const float* __restrict__ bpi,
                      const float* __restrict__ bmu,
                      const float* __restrict__ bsig,
                      float* __restrict__ out,
                      int Btot) {
  __shared__ unsigned short xs[BM][XS_STRIDE];                 // x tile, bf16
  __shared__ __align__(16) unsigned char pool[POOL_BYTES];     // unioned area

  unsigned short (*w1c)[W1_STRIDE] =
      reinterpret_cast<unsigned short(*)[W1_STRIDE]>(pool);                 // [64 cols][256 k]
  unsigned short (*whc)[WH_STRIDE] =
      reinterpret_cast<unsigned short(*)[WH_STRIDE]>(pool + 33792);         // [320 cols][64 k]
  unsigned short (*hbuf)[HB_STRIDE] =
      reinterpret_cast<unsigned short(*)[HB_STRIDE]>(pool + 33792 + 46080); // [128 rows][64 k]
  float (*stage)[324] = reinterpret_cast<float(*)[324]>(pool);              // [64 rows][320 cols]

  const int tid  = threadIdx.x;
  const int lane = tid & 31;
  const int w    = tid >> 5;            // wave id, 0..15
  const int row0 = blockIdx.x * BM;

  // ---- load x tile (f32 -> bf16), done once per block ----
  for (int i = tid; i < BM * BIN; i += NT) {
    int r = i >> 8, c = i & 255;
    xs[r][c] = f2bf(x[(size_t)(row0 + r) * BIN + c]);
  }

  // weight bytes for one chunk staged through VGPRs:
  //   w1c: 64*256 ushorts = 2048 uint4 -> 4 per thread
  //   whc: 320*64 ushorts = 2560 uint4 -> 5 per thread
  uint4 g1[4], g2[5];
  auto load_weights = [&](int ch) {
    const int hc0 = ch * HC;
#pragma unroll
    for (int it = 0; it < 4; ++it) {
      int i = tid + it * NT;           // [0, 2048)
      int c = i >> 5, q = i & 31;
      g1[it] = ((const uint4*)(W1t + (size_t)(hc0 + c) * BIN))[q];
    }
#pragma unroll
    for (int it = 0; it < 5; ++it) {
      int i = tid + it * NT;           // [0, 2560)
      int c = i >> 3, q = i & 7;
      g2[it] = ((const uint4*)(Wht + (size_t)c * BH + hc0))[q];
    }
  };
  auto store_weights = [&]() {
#pragma unroll
    for (int it = 0; it < 4; ++it) {
      int i = tid + it * NT;
      int c = i >> 5, q = i & 31;
      *(uint4*)&w1c[c][q * 8] = g1[it];
    }
#pragma unroll
    for (int it = 0; it < 5; ++it) {
      int i = tid + it * NT;
      int c = i >> 3, q = i & 7;
      *(uint4*)&whc[c][q * 8] = g2[it];
    }
  };

  // stage-2 persistent accumulators: wave w -> row slab (w&7), col tiles
  // (w>>3)*10 .. +9  (20 col tiles of 16 cover 320 columns)
  const int rs2   = (w & 7) * 16;
  const int cbase = (w >> 3) * 10;
  v8f Cacc[10];
#pragma unroll
  for (int j = 0; j < 10; ++j)
    Cacc[j] = (v8f){0.f, 0.f, 0.f, 0.f, 0.f, 0.f, 0.f, 0.f};

  // stage-1 assignment: tiles t=2w and 2w+1 share the column tile (same t>>3),
  // differ only in the batch-row slab -> shared B fragment, paired WMMAs.
  const int rA0  = ((2 * w) & 7) * 16;
  const int rA1  = rA0 + 16;
  const int c0s1 = (w >> 2) * 16;      // (2w)>>3 * 16

  load_weights(0);                      // preload chunk 0 into registers

  for (int ch = 0; ch < NCH; ++ch) {
    const int hc0 = ch * HC;
    __syncthreads();                    // prev stage-2 done reading whc/hbuf
    store_weights();                    // registers -> LDS (waits loadcnt here)
    if (ch + 1 < NCH) load_weights(ch + 1);   // issue next chunk's globals now
    __syncthreads();                    // w1c/whc ready

    // ---- stage 1: h[:, chunk] = tanh(x @ W1c + b1) ----
    {
      v8f acc0 = (v8f){0.f, 0.f, 0.f, 0.f, 0.f, 0.f, 0.f, 0.f};
      v8f acc1 = acc0;
#pragma unroll
      for (int kk = 0; kk < BIN / 32; ++kk) {
        v16bf b  = load_b(&w1c[0][0], W1_STRIDE, c0s1, kk * 32, lane);
        v16bf a0 = load_a(&xs[0][0],  XS_STRIDE, rA0,  kk * 32, lane);
        v16bf a1 = load_a(&xs[0][0],  XS_STRIDE, rA1,  kk * 32, lane);
        acc0 = wmma_bf16(a0, b, acc0);
        acc1 = wmma_bf16(a1, b, acc1);
      }
      const float bv = b1[hc0 + c0s1 + (lane & 15)];
      const int n = lane & 15, mb = (lane >> 4) * 8;
#pragma unroll
      for (int v = 0; v < 8; ++v) {
        hbuf[rA0 + mb + v][c0s1 + n] = f2bf(tanhf(acc0[v] + bv));
        hbuf[rA1 + mb + v][c0s1 + n] = f2bf(tanhf(acc1[v] + bv));
      }
    }
    __syncthreads();

    // ---- stage 2: Cacc += h_chunk @ Wh_chunk (paired B loads) ----
#pragma unroll
    for (int kk = 0; kk < 2; ++kk) {
      v16bf a = load_a(&hbuf[0][0], HB_STRIDE, rs2, kk * 32, lane);
#pragma unroll
      for (int j = 0; j < 10; j += 2) {
        v16bf b0 = load_b(&whc[0][0], WH_STRIDE, (cbase + j) * 16,     kk * 32, lane);
        v16bf b1 = load_b(&whc[0][0], WH_STRIDE, (cbase + j + 1) * 16, kk * 32, lane);
        Cacc[j]     = wmma_bf16(a, b0, Cacc[j]);
        Cacc[j + 1] = wmma_bf16(a, b1, Cacc[j + 1]);
      }
    }
  }

  // ---- epilogue in two 64-row passes through the staging buffer ----
  const long muOff  = (long)Btot * 32;
  const long sigOff = (long)Btot * 128;   // 32 + 96

  for (int pass = 0; pass < 2; ++pass) {
    __syncthreads();   // previous pass's reads (or last chunk's stage-2) done
    if (((w & 7) >> 2) == pass) {   // waves owning rows [pass*64, pass*64+64)
      const int n = lane & 15, mb = (lane >> 4) * 8;
      const int srow = rs2 - pass * 64;
#pragma unroll
      for (int j = 0; j < 10; ++j) {
        const int c0 = (cbase + j) * 16;
#pragma unroll
        for (int v = 0; v < 8; ++v)
          stage[srow + mb + v][c0 + n] = Cacc[j][v];
      }
    }
    __syncthreads();

    // 8 threads per batch row
    const int r  = tid >> 3;
    const int p  = tid & 7;
    const long gr = row0 + pass * 64 + r;
    const float* st = stage[r];

    if (p == 0) {   // softmax over 32 pi logits
      float vals[NK];
      float mx = -3.4e38f;
#pragma unroll
      for (int j = 0; j < NK; ++j) {
        vals[j] = st[j] + bpi[j];
        mx = fmaxf(mx, vals[j]);
      }
      float s = 0.f;
#pragma unroll
      for (int j = 0; j < NK; ++j) { vals[j] = expf(vals[j] - mx); s += vals[j]; }
      const float inv = 1.f / s;
#pragma unroll
      for (int j = 0; j < NK; ++j) out[gr * 32 + j] = vals[j] * inv;
    }

    // mu: 96 columns, 12 per helper thread
#pragma unroll
    for (int j = p * 12; j < p * 12 + 12; ++j)
      out[muOff + gr * 96 + j] = st[32 + j] + bmu[j];

    // sigma: 4 gaussians per helper thread
    for (int k = p * 4; k < p * 4 + 4; ++k) {
      const int sb = 128 + k * 6;
      float sv0 = st[sb + 0] + bsig[k * 6 + 0];
      float sv1 = st[sb + 1] + bsig[k * 6 + 1];
      float sv2 = st[sb + 2] + bsig[k * 6 + 2];
      float sv3 = st[sb + 3] + bsig[k * 6 + 3];
      float sv4 = st[sb + 4] + bsig[k * 6 + 4];
      float sv5 = st[sb + 5] + bsig[k * 6 + 5];
      float e0 = expf(sv0), e3 = expf(sv3), e5 = expf(sv5);
      float t1 = tanhf(sv1), t2 = tanhf(sv2), t4 = tanhf(sv4);
      float* o = out + sigOff + gr * 288 + (long)k * 9;
      o[0] = e0; o[1] = t1; o[2] = t2;
      o[3] = t1; o[4] = e3; o[5] = t4;
      o[6] = t2; o[7] = t4; o[8] = e5;
    }
  }
}

extern "C" void kernel_launch(void* const* d_in, const int* in_sizes, int n_in,
                              void* d_out, int out_size, void* d_ws, size_t ws_size,
                              hipStream_t stream) {
  const float* x    = (const float*)d_in[0];
  const float* W1   = (const float*)d_in[1];
  const float* b1   = (const float*)d_in[2];
  const float* Wpi  = (const float*)d_in[3];
  const float* bpi  = (const float*)d_in[4];
  const float* Wmu  = (const float*)d_in[5];
  const float* bmu  = (const float*)d_in[6];
  const float* Wsig = (const float*)d_in[7];
  const float* bsig = (const float*)d_in[8];
  float* out = (float*)d_out;
  unsigned short* ws = (unsigned short*)d_ws;

  const int Btot = in_sizes[0] / BIN;       // 131072

  // weights -> bf16 k-major in d_ws (1.15 MB)
  const int prepGrid = (WS_TOTAL + 255) / 256;
  mdn_prep_kernel<<<prepGrid, 256, 0, stream>>>(W1, Wpi, Wmu, Wsig, ws);

  const int grid = Btot / BM;               // 1024 blocks of 512 threads
  mdn_fused_kernel<<<grid, NT, 0, stream>>>(x, ws, ws + WS_W1T_ELEMS,
                                            b1, bpi, bmu, bsig, out, Btot);
}
